// PatchAttention_62818191671648
// MI455X (gfx1250) — matrix operand order repaired and checked by reading, not compiled, not run on hardware
//
#include <hip/hip_runtime.h>

typedef __attribute__((ext_vector_type(16))) _Float16 v16h;
typedef __attribute__((ext_vector_type(8)))  _Float16 v8h;
typedef __attribute__((ext_vector_type(8)))  float    v8f;

#define NB 4
#define NC 64
#define NH 48
#define NW 48
#define NHEADS 4
#define NCH 256          // NC * NHEADS
#define NHW 2304         // NH * NW
#define NBH 16           // NB * NHEADS

// ---------------------------------------------------------------------------
// K1: channel LayerNorm (Restormer WithBias) over NCHW channel dim
// ---------------------------------------------------------------------------
__global__ void k_layernorm(const float* __restrict__ x,
                            const float* __restrict__ w,
                            const float* __restrict__ b,
                            float* __restrict__ xn) {
  int idx = blockIdx.x * blockDim.x + threadIdx.x;   // over NB*NHW pixels
  if (idx >= NB * NHW) return;
  int bb = idx / NHW, n = idx % NHW;
  const float* xp = x + (size_t)bb * NC * NHW + n;
  float mu = 0.f;
  for (int c = 0; c < NC; ++c) mu += xp[(size_t)c * NHW];
  mu *= (1.0f / NC);
  float var = 0.f;
  for (int c = 0; c < NC; ++c) { float d = xp[(size_t)c * NHW] - mu; var += d * d; }
  var *= (1.0f / NC);
  float inv = rsqrtf(var + 1e-5f);
  float* op = xn + (size_t)bb * NC * NHW + n;
  for (int c = 0; c < NC; ++c) op[(size_t)c * NHW] = (xp[(size_t)c * NHW] - mu) * inv * w[c] + b[c];
}

// ---------------------------------------------------------------------------
// K2: 1x1 conv (no bias):  y[b][o][n] = sum_c W[o][c] * xn[b][c][n]
// ---------------------------------------------------------------------------
__global__ void k_conv1x1(const float* __restrict__ xn,
                          const float* __restrict__ wgt,
                          float* __restrict__ y) {
  int idx = blockIdx.x * blockDim.x + threadIdx.x;   // NB*NCH*NHW, b-major
  if (idx >= NB * NCH * NHW) return;
  int n = idx % NHW; int t = idx / NHW; int o = t % NCH; int bb = t / NCH;
  const float* xp = xn + (size_t)bb * NC * NHW + n;
  const float* wp = wgt + (size_t)o * NC;
  float acc = 0.f;
  for (int c = 0; c < NC; ++c) acc += wp[c] * xp[(size_t)c * NHW];
  y[idx] = acc;
}

// ---------------------------------------------------------------------------
// K3: depthwise 3x3 (pad=1) + transpose to (b, head, n, c) + optional L2-norm
// ---------------------------------------------------------------------------
__global__ void k_dwconv_norm(const float* __restrict__ y,
                              const float* __restrict__ dw,
                              _Float16* __restrict__ out,
                              int do_norm) {
  __shared__ float red[64];
  int n  = blockIdx.x;                 // spatial index 0..NHW-1
  int bh = blockIdx.y;                 // b*NHEADS + head
  int bb = bh / NHEADS, head = bh % NHEADS;
  int c  = threadIdx.x;                // 0..63
  int ch = head * NC + c;              // channel in (head c) layout
  int hh = n / NW, ww = n % NW;
  const float* yp = y + ((size_t)bb * NCH + ch) * NHW;
  const float* kp = dw + (size_t)ch * 9;
  float acc = 0.f;
  #pragma unroll
  for (int dy = -1; dy <= 1; ++dy) {
    int yy = hh + dy; if (yy < 0 || yy >= NH) continue;
    #pragma unroll
    for (int dx = -1; dx <= 1; ++dx) {
      int xx = ww + dx; if (xx < 0 || xx >= NW) continue;
      acc += kp[(dy + 1) * 3 + (dx + 1)] * yp[yy * NW + xx];
    }
  }
  float val = acc;
  if (do_norm) {
    red[c] = val * val;
    __syncthreads();
    for (int s = 32; s > 0; s >>= 1) { if (c < s) red[c] += red[c + s]; __syncthreads(); }
    float inv = 1.0f / fmaxf(sqrtf(red[0]), 1e-12f);
    val *= inv;
  }
  out[(((size_t)bh) * NHW + n) * NC + c] = (_Float16)val;
}

// ---------------------------------------------------------------------------
// K4: InstanceNorm statistics of S = QK^T/sqrt(N) without forming S:
//     sum(S)  = (sum_n q_n) . (sum_m k_m) / sqrt(N)
//     sum(S^2)= (sum_n q_n^T G q_n) / N,  G = sum_m k_m k_m^T   (64x64)
// ---------------------------------------------------------------------------
__global__ void k_stats(const _Float16* __restrict__ qh,
                        const _Float16* __restrict__ kh,
                        float* __restrict__ stats) {
  __shared__ float Gs[64 * 64];
  __shared__ float sq[64], sk[64];
  __shared__ float redsum[256];
  int bh = blockIdx.x;
  const _Float16* qp = qh + (size_t)bh * NHW * NC;
  const _Float16* kp = kh + (size_t)bh * NHW * NC;
  int t = threadIdx.x;
  if (t < 64) {
    float aq = 0.f, ak = 0.f;
    for (int m = 0; m < NHW; ++m) { aq += (float)qp[(size_t)m * NC + t]; ak += (float)kp[(size_t)m * NC + t]; }
    sq[t] = aq; sk[t] = ak;
  }
  for (int e = t; e < 4096; e += 256) {
    int c1 = e >> 6, c2 = e & 63;
    float g = 0.f;
    for (int m = 0; m < NHW; ++m) g += (float)kp[(size_t)m * NC + c1] * (float)kp[(size_t)m * NC + c2];
    Gs[e] = g;
  }
  __syncthreads();
  float qgq = 0.f;
  for (int m = t; m < NHW; m += 256) {
    float qv[64];
    for (int c = 0; c < 64; ++c) qv[c] = (float)qp[(size_t)m * NC + c];
    for (int c1 = 0; c1 < 64; ++c1) {
      float inner = 0.f;
      for (int c2 = 0; c2 < 64; ++c2) inner += Gs[c1 * 64 + c2] * qv[c2];
      qgq += qv[c1] * inner;
    }
  }
  redsum[t] = qgq;
  __syncthreads();
  for (int s = 128; s > 0; s >>= 1) { if (t < s) redsum[t] += redsum[t + s]; __syncthreads(); }
  if (t == 0) {
    float dqk = 0.f;
    for (int c = 0; c < 64; ++c) dqk += sq[c] * sk[c];
    float num = (float)NHW;
    float mean = dqk * rsqrtf(num) / (num * num);
    float es2  = redsum[0] / (num * num * num);
    float var  = es2 - mean * mean;
    stats[bh * 2 + 0] = mean;
    stats[bh * 2 + 1] = rsqrtf(var + 1e-5f);
  }
}

// ---------------------------------------------------------------------------
// K5: fused single-pass attention, 32 query rows per wave (two A row-tiles
//     sharing every K/V B-operand and the LDS V-transpose staging).
//     Softmax shift uses the analytic bound from L2-normalized Q,K
//     (|q.k| <= ~1), so no row-max pass is needed.  The shifted exp argument
//     is provably in [-20, 0], so the raw hardware v_exp_f32
//     (__builtin_amdgcn_exp2f) is exact -- no libm range-fixup code.
// ---------------------------------------------------------------------------
__device__ inline v16h load_a_f16(const _Float16* p0, const _Float16* p1) {
  v8h lo = *(const v8h*)p0;
  v8h hi = *(const v8h*)p1;
  v16h a;
  #pragma unroll
  for (int i = 0; i < 8; ++i) { a[i] = lo[i]; a[i + 8] = hi[i]; }
  return a;
}

__global__ void __launch_bounds__(32)
k_attention(const _Float16* __restrict__ qh, const _Float16* __restrict__ kh,
            const _Float16* __restrict__ vh, const float* __restrict__ stats,
            float* __restrict__ oh) {
  __shared__ __attribute__((aligned(32))) _Float16 Vt[64 * 32]; // V chunk, (c, m)
  __shared__ __attribute__((aligned(32))) _Float16 Pl[32 * 32]; // P tiles, (row, m)
  int rb = blockIdx.x;            // 32-row block, 0..71
  int bh = blockIdx.y;            // (b, head)
  int lane = threadIdx.x;
  int r = lane & 15, sel = lane >> 4;
  const _Float16* qp = qh + (size_t)bh * NHW * NC;
  const _Float16* kp = kh + (size_t)bh * NHW * NC;
  const _Float16* vp = vh + (size_t)bh * NHW * NC;
  float mean = stats[bh * 2 + 0], invstd = stats[bh * 2 + 1];
  const float LOG2E = 1.44269504f;
  float inv_s = rsqrtf((float)NHW);
  float za = inv_s * invstd;
  float zb = -mean * invstd;
  float z_ub = 1.01f * za + zb;            // |dot| <= ~1 (f16-rounded unit vectors)
  float ea = za * LOG2E;
  float eb = (zb - z_ub) * LOG2E;          // exp(z - z_ub) == exp2(dot*ea + eb)
  int n0 = rb * 32;

  // A operands: two Q row-tiles (rows n0..n0+15, n0+16..n0+31), K in [0,32),[32,64)
  v16h aq[2][2];
  #pragma unroll
  for (int rt = 0; rt < 2; ++rt) {
    const _Float16* qrow = qp + (size_t)(n0 + rt * 16 + r) * NC;
    aq[rt][0] = load_a_f16(qrow + 0  + sel * 16, qrow + 8  + sel * 16);
    aq[rt][1] = load_a_f16(qrow + 32 + sel * 16, qrow + 40 + sel * 16);
  }

  float dsum[2][8];
  #pragma unroll
  for (int rt = 0; rt < 2; ++rt)
    #pragma unroll
    for (int i = 0; i < 8; ++i) dsum[rt][i] = 0.f;
  v8f oacc[2][4];
  #pragma unroll
  for (int rt = 0; rt < 2; ++rt)
    #pragma unroll
    for (int g = 0; g < 4; ++g) oacc[rt][g] = (v8f){};

  for (int m0 = 0; m0 < NHW; m0 += 32) {
    { // stage V[m0..m0+31][0..63] transposed into LDS (c-major)
      const _Float16* vr = vp + (size_t)(m0 + lane) * NC;
      #pragma unroll
      for (int c = 0; c < 64; ++c) Vt[c * 32 + lane] = vr[c];
    }
    if (m0 + 32 < NHW) {  // prefetch next K/V chunk (global_prefetch_b8)
      __builtin_prefetch(kp + (size_t)(m0 + 32 + lane) * NC, 0, 1);
      __builtin_prefetch(vp + (size_t)(m0 + 32 + lane) * NC, 0, 1);
    }
    __syncthreads();
    #pragma unroll
    for (int t = 0; t < 2; ++t) {
      const _Float16* kr = kp + (size_t)(m0 + t * 16 + r) * NC;
      v16h b0 = *(const v16h*)(kr + 0  + sel * 16);   // shared by both row-tiles
      v16h b1 = *(const v16h*)(kr + 32 + sel * 16);
      #pragma unroll
      for (int rt = 0; rt < 2; ++rt) {
        v8f acc = {};
        acc = __builtin_amdgcn_wmma_f32_16x16x32_f16(false, aq[rt][0], false, b0, (short)0, acc, false, false);
        acc = __builtin_amdgcn_wmma_f32_16x16x32_f16(false, aq[rt][1], false, b1, (short)0, acc, false, false);
        #pragma unroll
        for (int i = 0; i < 8; ++i) {
          float p = __builtin_amdgcn_exp2f(acc[i] * ea + eb);  // raw v_exp_f32
          dsum[rt][i] += p;
          Pl[(rt * 16 + i + sel * 8) * 32 + t * 16 + r] = (_Float16)p;
        }
      }
    }
    __syncthreads();
    // O(32x64) += P(32x32) * V(32x64): V B-operands shared by both row-tiles
    {
      v16h bv0 = *(const v16h*)(&Vt[( 0 + r) * 32 + sel * 16]);
      v16h bv1 = *(const v16h*)(&Vt[(16 + r) * 32 + sel * 16]);
      v16h bv2 = *(const v16h*)(&Vt[(32 + r) * 32 + sel * 16]);
      v16h bv3 = *(const v16h*)(&Vt[(48 + r) * 32 + sel * 16]);
      #pragma unroll
      for (int rt = 0; rt < 2; ++rt) {
        const _Float16* prow = &Pl[(rt * 16 + r) * 32];
        v16h ap = load_a_f16(prow + sel * 16, prow + 8 + sel * 16);
        oacc[rt][0] = __builtin_amdgcn_wmma_f32_16x16x32_f16(false, ap, false, bv0, (short)0, oacc[rt][0], false, false);
        oacc[rt][1] = __builtin_amdgcn_wmma_f32_16x16x32_f16(false, ap, false, bv1, (short)0, oacc[rt][1], false, false);
        oacc[rt][2] = __builtin_amdgcn_wmma_f32_16x16x32_f16(false, ap, false, bv2, (short)0, oacc[rt][2], false, false);
        oacc[rt][3] = __builtin_amdgcn_wmma_f32_16x16x32_f16(false, ap, false, bv3, (short)0, oacc[rt][3], false, false);
      }
    }
    __syncthreads();
  }

  #pragma unroll
  for (int m = 1; m < 16; m <<= 1)
    #pragma unroll
    for (int rt = 0; rt < 2; ++rt)
      #pragma unroll
      for (int i = 0; i < 8; ++i)
        dsum[rt][i] += __shfl_xor(dsum[rt][i], m, 32);

  float* op = oh + ((size_t)bh * NHW + n0) * NC;
  #pragma unroll
  for (int rt = 0; rt < 2; ++rt)
    #pragma unroll
    for (int i = 0; i < 8; ++i) {
      int row = rt * 16 + i + sel * 8;
      float inv = __builtin_amdgcn_rcpf(dsum[rt][i]);
      op[(size_t)row * NC +  0 + r] = oacc[rt][0][i] * inv;
      op[(size_t)row * NC + 16 + r] = oacc[rt][1][i] * inv;
      op[(size_t)row * NC + 32 + r] = oacc[rt][2][i] * inv;
      op[(size_t)row * NC + 48 + r] = oacc[rt][3][i] * inv;
    }
}

// ---------------------------------------------------------------------------
// K6: head-mean + 1x1 projection + bias + residual
// ---------------------------------------------------------------------------
__global__ void k_proj(const float* __restrict__ oh, const float* __restrict__ pw,
                       const float* __restrict__ pb, const float* __restrict__ org,
                       float* __restrict__ out) {
  int idx = blockIdx.x * blockDim.x + threadIdx.x;  // NB*NC*NHW, (b,o,n)
  if (idx >= NB * NC * NHW) return;
  int n = idx % NHW; int t = idx / NHW; int o = t % NC; int bb = t / NC;
  const float* ohp = oh + (size_t)bb * NHEADS * NHW * NC + (size_t)n * NC;
  const float* wp = pw + (size_t)o * NC;
  float acc = 0.f;
  for (int c = 0; c < NC; ++c) {
    float mc = 0.25f * (ohp[c] + ohp[(size_t)NHW * NC + c] +
                        ohp[2 * (size_t)NHW * NC + c] + ohp[3 * (size_t)NHW * NC + c]);
    acc += wp[c] * mc;
  }
  out[idx] = org[idx] + acc + pb[o];
}

// ---------------------------------------------------------------------------
extern "C" void kernel_launch(void* const* d_in, const int* in_sizes, int n_in,
                              void* d_out, int out_size, void* d_ws, size_t ws_size,
                              hipStream_t stream) {
  const float* emb  = (const float*)d_in[0];
  const float* ln_w = (const float*)d_in[1];
  const float* ln_b = (const float*)d_in[2];
  const float* w1[3]  = { (const float*)d_in[3], (const float*)d_in[4], (const float*)d_in[5] };
  const float* dws[3] = { (const float*)d_in[6], (const float*)d_in[7], (const float*)d_in[8] };
  const float* pw = (const float*)d_in[9];
  const float* pb = (const float*)d_in[10];
  float* out = (float*)d_out;

  // workspace layout (all offsets 256B aligned)
  const size_t XLN_F = (size_t)NB * NC * NHW;        // 589824 floats
  const size_t YB_F  = (size_t)NB * NCH * NHW;       // 2359296 floats
  const size_t QH_H  = (size_t)NBH * NHW * NC;       // 2359296 halves

  char* ws = (char*)d_ws;
  float*    xln   = (float*)ws;                         ws += XLN_F * sizeof(float);
  float*    ybuf  = (float*)ws;                         ws += YB_F  * sizeof(float);
  _Float16* qkvh[3];
  qkvh[0] = (_Float16*)ws;                              ws += QH_H * sizeof(_Float16);
  qkvh[1] = (_Float16*)ws;                              ws += QH_H * sizeof(_Float16);
  qkvh[2] = (_Float16*)ws;                              ws += QH_H * sizeof(_Float16);
  float*    stats = (float*)ws;                         ws += 64 * sizeof(float);
  float*    oh    = (float*)ws;                         /* QH_H floats */

  // K1: layernorm
  k_layernorm<<<(NB * NHW + 255) / 256, 256, 0, stream>>>(emb, ln_w, ln_b, xln);

  // K2+K3 per branch (ybuf reused serially; stream ordering guarantees safety)
  for (int br = 0; br < 3; ++br) {
    k_conv1x1<<<(NB * NCH * NHW + 255) / 256, 256, 0, stream>>>(xln, w1[br], ybuf);
    k_dwconv_norm<<<dim3(NHW, NBH), 64, 0, stream>>>(ybuf, dws[br], qkvh[br], br < 2 ? 1 : 0);
  }

  // K4: instance-norm statistics via G = K^T K trick
  k_stats<<<NBH, 256, 0, stream>>>(qkvh[0], qkvh[1], stats);

  // K5: fused single-pass WMMA attention (one wave per 32-row block)
  k_attention<<<dim3(NHW / 32, NBH), 32, 0, stream>>>(qkvh[0], qkvh[1], qkvh[2], stats, oh);

  // K6: head mean + projection + residual
  k_proj<<<(NB * NC * NHW + 255) / 256, 256, 0, stream>>>(oh, pw, pb, emb, out);
}
